// MPNNet_6408091205706
// MI455X (gfx1250) — matrix operand order repaired and checked by reading, not compiled, not run on hardware
//
#include <hip/hip_runtime.h>
#include <hip/hip_bf16.h>
#include <stdint.h>

#define DIM      64
#define N_GRAPHS 300
#define N_NODES  12000
#define N_EDGES  40000

typedef __attribute__((ext_vector_type(16))) _Float16 v16h;
typedef __attribute__((ext_vector_type(8)))  _Float16 v8h;
typedef __attribute__((ext_vector_type(2)))  _Float16 v2h;
typedef __attribute__((ext_vector_type(8)))  float    v8f;

__device__ __forceinline__ float sigmoidf_(float x) { return 1.0f / (1.0f + expf(-x)); }

// monotonic float<->uint encoding for atomicMax on floats (handles negatives)
__device__ __forceinline__ unsigned encf(float x) {
    unsigned b = __float_as_uint(x);
    return (b & 0x80000000u) ? ~b : (b | 0x80000000u);
}
__device__ __forceinline__ float decf(unsigned k) {
    return (k & 0x80000000u) ? __uint_as_float(k & 0x7FFFFFFFu) : __uint_as_float(~k);
}

// ---------------------------------------------------------------------------
// Edge-MLP layer-2 GEMM, all-f16 inputs, 4x N-tile blocking per wave.
//   We[M,Nn] = A[M,K](f16) * B[Nn,K](f16)^T + bias[n]   (f16 out)
// A fragment loaded once per K-chunk and reused across 4 WMMAs.
// M mult of 16, Nn mult of 64, K mult of 32.
// ---------------------------------------------------------------------------
template <int K>
__global__ void wmma_gemm_f16_kernel(const _Float16* __restrict__ A,
                                     const _Float16* __restrict__ B,
                                     const float* __restrict__ bias,
                                     _Float16* __restrict__ C, int M, int Nn)
{
    const int waveId = blockIdx.x * (blockDim.x >> 5) + (threadIdx.x >> 5);
    const int NtQ = Nn >> 6;                 // groups of 4 n-tiles
    const int Mt  = M >> 4;
    if (waveId >= Mt * NtQ) return;
    const int mt  = waveId / NtQ;
    const int ntQ = waveId % NtQ;
    const int lane = threadIdx.x & 31;
    const int hsel = lane >> 4;
    const int l16  = lane & 15;

    const _Float16* Arow = A + (size_t)(mt * 16 + l16) * K;

    v8f acc[4] = {{}, {}, {}, {}};
#pragma unroll
    for (int k0 = 0; k0 < K; k0 += 32) {
        union { v16h v; v8h h[2]; } au;
        const int ka = k0 + hsel * 8;        // A K {ka..ka+7, ka+16..ka+23}
        au.h[0] = *(const v8h*)(Arow + ka);
        au.h[1] = *(const v8h*)(Arow + ka + 16);
        const int kb = k0 + hsel * 8;        // B K {kb..kb+7, kb+16..kb+23}
#pragma unroll
        for (int j = 0; j < 4; ++j) {
            const _Float16* Brow = B + (size_t)((ntQ * 4 + j) * 16 + l16) * K;
            union { v16h v; v8h h[2]; } bu;
            bu.h[0] = *(const v8h*)(Brow + kb);
            bu.h[1] = *(const v8h*)(Brow + kb + 16);
            acc[j] = __builtin_amdgcn_wmma_f32_16x16x32_f16(false, au.v, false, bu.v,
                                                            (short)0, acc[j], false, false);
        }
    }
#pragma unroll
    for (int j = 0; j < 4; ++j) {
        const int n = (ntQ * 4 + j) * 16 + l16;
        const float bv = bias[n];
#pragma unroll
        for (int v = 0; v < 8; ++v) {
            const int m = mt * 16 + hsel * 8 + v;
            C[(size_t)m * Nn + n] = (_Float16)(acc[j][v] + bv);
        }
    }
}

// ---------------------------------------------------------------------------
// Generic WMMA GEMM (f32 in/out, convert-on-load):
//   C[M,N] = act( A[M,K] * B[N,K]^T + bias[n] + rowScale[m]*Cadd[m,n] )
// ---------------------------------------------------------------------------
template <int K>
__global__ void wmma_gemm_f32_kernel(const float* __restrict__ A, const float* __restrict__ B,
                                     const float* __restrict__ bias, const float* __restrict__ Cadd,
                                     const float* __restrict__ rowScale, float* __restrict__ C,
                                     int M, int N, int relu)
{
    const int waveId = blockIdx.x * (blockDim.x >> 5) + (threadIdx.x >> 5);
    const int Nt = N >> 4;
    const int Mt = M >> 4;
    if (waveId >= Mt * Nt) return;
    const int mt = waveId / Nt;
    const int nt = waveId % Nt;
    const int lane = threadIdx.x & 31;
    const int hsel = lane >> 4;
    const int l16  = lane & 15;

    const float* Arow = A + (size_t)(mt * 16 + l16) * K;
    const float* Brow = B + (size_t)(nt * 16 + l16) * K;

    v8f acc = {};
#pragma unroll
    for (int k0 = 0; k0 < K; k0 += 32) {
        v16h a, b;
        const int ka = k0 + hsel * 8;
#pragma unroll
        for (int i = 0; i < 8; ++i) {
            a[i]     = (_Float16)Arow[ka + i];
            a[i + 8] = (_Float16)Arow[ka + 16 + i];
        }
        const int kb = k0 + hsel * 8;
#pragma unroll
        for (int i = 0; i < 8; ++i) {
            b[i]     = (_Float16)Brow[kb + i];
            b[i + 8] = (_Float16)Brow[kb + 16 + i];
        }
        acc = __builtin_amdgcn_wmma_f32_16x16x32_f16(false, a, false, b,
                                                     (short)0, acc, false, false);
    }
    const int n = nt * 16 + l16;
    const float bv = bias ? bias[n] : 0.0f;
#pragma unroll
    for (int v = 0; v < 8; ++v) {
        const int m = mt * 16 + hsel * 8 + v;
        float val = acc[v] + bv;
        if (Cadd) {
            const float s = rowScale ? rowScale[m] : 1.0f;
            val += s * Cadd[(size_t)m * N + n];
        }
        if (relu) val = fmaxf(val, 0.0f);
        C[(size_t)m * N + n] = val;
    }
}

// lin0: out[n,d] = relu( sum_k x[n,k]*lin0_w[d,k] + lin0_b[d] ), K=14
__global__ void lin0_kernel(const float* __restrict__ x, const float* __restrict__ w,
                            const float* __restrict__ b, float* __restrict__ out)
{
    int idx = blockIdx.x * blockDim.x + threadIdx.x;
    if (idx >= N_NODES * DIM) return;
    int n = idx >> 6, d = idx & 63;
    float v = b[d];
    const float* xr = x + (size_t)n * 14;
    const float* wr = w + (size_t)d * 14;
#pragma unroll
    for (int k = 0; k < 14; ++k) v += xr[k] * wr[k];
    out[idx] = fmaxf(v, 0.0f);
}

// edge MLP layer 1 -> f16 hidden: hid[e,j] = relu( sum_k ea[e,k]*nn1_w[j,k] + nn1_b[j] )
__global__ void edge_hidden_kernel(const float* __restrict__ ea, const float* __restrict__ w,
                                   const float* __restrict__ b, _Float16* __restrict__ hid)
{
    int idx = blockIdx.x * blockDim.x + threadIdx.x;
    if (idx >= N_EDGES * 128) return;
    int e = idx >> 7, j = idx & 127;
    const float* er = ea + (size_t)e * 4;
    const float* wr = w + (size_t)j * 4;
    float v = b[j] + er[0] * wr[0] + er[1] * wr[1] + er[2] * wr[2] + er[3] * wr[3];
    hid[idx] = (_Float16)fmaxf(v, 0.0f);
}

__global__ void cvt_f32_to_f16_kernel(const float* __restrict__ src, _Float16* __restrict__ dst,
                                      int n)
{
    int i = blockIdx.x * blockDim.x + threadIdx.x;
    if (i < n) dst[i] = (_Float16)src[i];
}

__global__ void transpose64_kernel(const float* __restrict__ w, float* __restrict__ wt)
{
    int i = blockIdx.x * blockDim.x + threadIdx.x;
    if (i >= 64 * 64) return;
    int k = i >> 6, o = i & 63;
    wt[(size_t)o * 64 + k] = w[(size_t)k * 64 + o];
}

__global__ void fill_f32_kernel(float* p, float v, int n)
{
    int i = blockIdx.x * blockDim.x + threadIdx.x;
    if (i < n) p[i] = v;
}
__global__ void fill_u32_kernel(unsigned* p, unsigned v, int n)
{
    int i = blockIdx.x * blockDim.x + threadIdx.x;
    if (i < n) p[i] = v;
}

__global__ void deg_count_kernel(const int* __restrict__ ei, float* __restrict__ deg)
{
    int e = blockIdx.x * blockDim.x + threadIdx.x;
    if (e < N_EDGES) atomicAdd(&deg[ei[N_EDGES + e]], 1.0f);
}
__global__ void deginv_kernel(const float* __restrict__ deg, float* __restrict__ dinv)
{
    int i = blockIdx.x * blockDim.x + threadIdx.x;
    if (i < N_NODES) dinv[i] = 1.0f / fmaxf(deg[i], 1.0f);
}

// ---------------------------------------------------------------------------
// Per-edge bmm + scatter: msg[e,o] = sum_i out[src[e],i]*We[e,i,o]; agg[dst]+=msg
// One wave per edge; We in f16 (halves the dominant HBM stream); source row
// broadcast via __shfl.
// ---------------------------------------------------------------------------
__global__ void edge_msg_kernel(const _Float16* __restrict__ We, const float* __restrict__ out,
                                const int* __restrict__ ei, float* __restrict__ agg)
{
    int e = blockIdx.x * (blockDim.x >> 5) + (threadIdx.x >> 5);
    if (e >= N_EDGES) return;
    const int lane = threadIdx.x & 31;
    const int src = ei[e];
    const int dst = ei[N_EDGES + e];

    const float s0 = out[(size_t)src * 64 + lane];
    const float s1 = out[(size_t)src * 64 + lane + 32];

    const _Float16* W = We + (size_t)e * 4096;
    float a0 = 0.0f, a1 = 0.0f;
#pragma unroll 4
    for (int i = 0; i < 64; ++i) {
        const float si = (i < 32) ? __shfl(s0, i, 32) : __shfl(s1, i - 32, 32);
        const v2h wv = *(const v2h*)(W + (size_t)i * 64 + 2 * lane);
        a0 += si * (float)wv[0];
        a1 += si * (float)wv[1];
    }
    atomicAdd(&agg[(size_t)dst * 64 + 2 * lane],     a0);
    atomicAdd(&agg[(size_t)dst * 64 + 2 * lane + 1], a1);
}

// GRU elementwise: h = (1-z)*n + z*h (torch gate order r,z,n)
__global__ void gru_kernel(const float* __restrict__ gx, const float* __restrict__ gh,
                           float* __restrict__ h)
{
    int idx = blockIdx.x * blockDim.x + threadIdx.x;
    if (idx >= N_NODES * DIM) return;
    int n = idx >> 6, d = idx & 63;
    const float* x  = gx + (size_t)n * 192;
    const float* hh = gh + (size_t)n * 192;
    float r  = sigmoidf_(x[d] + hh[d]);
    float z  = sigmoidf_(x[64 + d] + hh[64 + d]);
    float nn = tanhf(x[128 + d] + r * hh[128 + d]);
    h[idx] = (1.0f - z) * nn + z * h[idx];
}

// Set2Set LSTM cell, one block (64 threads) per graph
__global__ void lstm_cell_kernel(const float* __restrict__ qstar, float* __restrict__ hs,
                                 float* __restrict__ cs,
                                 const float* __restrict__ w_ih, const float* __restrict__ w_hh,
                                 const float* __restrict__ b_ih, const float* __restrict__ b_hh)
{
    __shared__ float sq[128];
    __shared__ float sh[64];
    const int b = blockIdx.x, d = threadIdx.x;
    sq[d]      = qstar[(size_t)b * 128 + d];
    sq[d + 64] = qstar[(size_t)b * 128 + 64 + d];
    sh[d]      = hs[(size_t)b * 64 + d];
    __syncthreads();
    float g[4];
#pragma unroll
    for (int gi = 0; gi < 4; ++gi) {
        const int row = gi * 64 + d;
        float v = b_ih[row] + b_hh[row];
        const float* wi = w_ih + (size_t)row * 128;
        const float* wh = w_hh + (size_t)row * 64;
        for (int k = 0; k < 128; ++k) v += sq[k] * wi[k];
        for (int k = 0; k < 64; ++k)  v += sh[k] * wh[k];
        g[gi] = v;
    }
    const float ig = sigmoidf_(g[0]), fg = sigmoidf_(g[1]);
    const float gg = tanhf(g[2]),     og = sigmoidf_(g[3]);
    const float c = fg * cs[(size_t)b * 64 + d] + ig * gg;
    cs[(size_t)b * 64 + d] = c;
    hs[(size_t)b * 64 + d] = og * tanhf(c);
}

__global__ void attn_dot_kernel(const float* __restrict__ out, const float* __restrict__ hs,
                                const int* __restrict__ batch, float* __restrict__ ebuf,
                                unsigned* __restrict__ emaxu)
{
    int n = blockIdx.x * blockDim.x + threadIdx.x;
    if (n >= N_NODES) return;
    const int b = batch[n];
    const float* o = out + (size_t)n * 64;
    const float* q = hs + (size_t)b * 64;
    float e = 0.0f;
#pragma unroll
    for (int d = 0; d < 64; ++d) e += o[d] * q[d];
    ebuf[n] = e;
    atomicMax(&emaxu[b], encf(e));
}

__global__ void attn_exp_kernel(float* __restrict__ ebuf, const unsigned* __restrict__ emaxu,
                                const int* __restrict__ batch, float* __restrict__ asum)
{
    int n = blockIdx.x * blockDim.x + threadIdx.x;
    if (n >= N_NODES) return;
    const int b = batch[n];
    const float a = expf(ebuf[n] - decf(emaxu[b]));
    ebuf[n] = a;
    atomicAdd(&asum[b], a);
}

__global__ void attn_scatter_kernel(const float* __restrict__ ebuf, const float* __restrict__ asum,
                                    const float* __restrict__ out, const int* __restrict__ batch,
                                    float* __restrict__ rread)
{
    int idx = blockIdx.x * blockDim.x + threadIdx.x;
    if (idx >= N_NODES * DIM) return;
    int n = idx >> 6, d = idx & 63;
    const int b = batch[n];
    const float w = ebuf[n] / asum[b];
    atomicAdd(&rread[(size_t)b * 64 + d], w * out[idx]);
}

__global__ void qstar_update_kernel(const float* __restrict__ hs, const float* __restrict__ rread,
                                    float* __restrict__ qstar)
{
    int idx = blockIdx.x * blockDim.x + threadIdx.x;
    if (idx >= N_GRAPHS * DIM) return;
    int b = idx >> 6, d = idx & 63;
    qstar[(size_t)b * 128 + d]      = hs[idx];
    qstar[(size_t)b * 128 + 64 + d] = rread[idx];
}

__global__ void readout_kernel(const float* __restrict__ qstar, const float* __restrict__ l1w,
                               const float* __restrict__ l1b, const float* __restrict__ l2w,
                               const float* __restrict__ l2b, float* __restrict__ y)
{
    __shared__ float sq[128];
    __shared__ float red[64];
    const int b = blockIdx.x, t = threadIdx.x;
    sq[t]      = qstar[(size_t)b * 128 + t];
    sq[t + 64] = qstar[(size_t)b * 128 + 64 + t];
    __syncthreads();
    float v = l1b[t];
    const float* w = l1w + (size_t)t * 128;
    for (int k = 0; k < 128; ++k) v += sq[k] * w[k];
    red[t] = fmaxf(v, 0.0f) * l2w[t];
    __syncthreads();
    for (int s = 32; s > 0; s >>= 1) {
        if (t < s) red[t] += red[t + s];
        __syncthreads();
    }
    if (t == 0) y[b] = red[0] + l2b[0];
}

// ---------------------------------------------------------------------------
extern "C" void kernel_launch(void* const* d_in, const int* in_sizes, int n_in,
                              void* d_out, int out_size, void* d_ws, size_t ws_size,
                              hipStream_t stream)
{
    const float* x        = (const float*)d_in[0];
    const int*   ei       = (const int*)d_in[1];
    const float* eattr    = (const float*)d_in[2];
    const int*   batch    = (const int*)d_in[3];
    const float* lin0_w   = (const float*)d_in[4];
    const float* lin0_b   = (const float*)d_in[5];
    const float* nn1_w    = (const float*)d_in[6];
    const float* nn1_b    = (const float*)d_in[7];
    const float* nn2_w    = (const float*)d_in[8];
    const float* nn2_b    = (const float*)d_in[9];
    const float* root_w   = (const float*)d_in[10];
    const float* conv_b   = (const float*)d_in[11];
    const float* gru_w_ih = (const float*)d_in[12];
    const float* gru_w_hh = (const float*)d_in[13];
    const float* gru_b_ih = (const float*)d_in[14];
    const float* gru_b_hh = (const float*)d_in[15];
    const float* lstm_w_ih= (const float*)d_in[16];
    const float* lstm_w_hh= (const float*)d_in[17];
    const float* lstm_b_ih= (const float*)d_in[18];
    const float* lstm_b_hh= (const float*)d_in[19];
    const float* lin1_w   = (const float*)d_in[20];
    const float* lin1_b   = (const float*)d_in[21];
    const float* lin2_w   = (const float*)d_in[22];
    const float* lin2_b   = (const float*)d_in[23];

    const int N = N_NODES, E = N_EDGES, B = N_GRAPHS;

    // ---- workspace layout (256B aligned slots) ----
    char* ws = (char*)d_ws;
    size_t off = 0;
    auto take = [&](size_t bytes) -> char* {
        char* p = ws + off;
        off += (bytes + 255) & ~(size_t)255;
        return p;
    };
    _Float16* We      = (_Float16*)take((size_t)E * 4096 * sizeof(_Float16)); // 328 MB
    _Float16* hidden_h= (_Float16*)take((size_t)E * 128 * sizeof(_Float16));  // 10 MB
    _Float16* nn2w_h  = (_Float16*)take((size_t)4096 * 128 * sizeof(_Float16));
    float* h          = (float*)take((size_t)N * 64 * 4);   // node state (== out)
    float* mbuf       = (float*)take((size_t)N * 64 * 4);
    float* agg        = (float*)take((size_t)N * 64 * 4);
    float* gx         = (float*)take((size_t)N * 192 * 4);
    float* gh         = (float*)take((size_t)N * 192 * 4);
    float* degbuf     = (float*)take((size_t)N * 4);
    float* deginv     = (float*)take((size_t)N * 4);
    float* rootT      = (float*)take(64 * 64 * 4);
    float* qstar      = (float*)take((size_t)B * 128 * 4);
    float* hs         = (float*)take((size_t)B * 64 * 4);
    float* cs         = (float*)take((size_t)B * 64 * 4);
    float* rread      = (float*)take((size_t)B * 64 * 4);
    float* ebuf       = (float*)take((size_t)N * 4);
    unsigned* emaxu   = (unsigned*)take((size_t)B * 4);
    float* asum       = (float*)take((size_t)B * 4);

    auto blks = [](int n, int t) { return (n + t - 1) / t; };

    // 1) lin0 -> h
    lin0_kernel<<<blks(N * 64, 256), 256, 0, stream>>>(x, lin0_w, lin0_b, h);

    // 2) edge MLP layer1 -> hidden (f16) ; nn2_w -> f16 copy
    edge_hidden_kernel<<<blks(E * 128, 256), 256, 0, stream>>>(eattr, nn1_w, nn1_b, hidden_h);
    cvt_f32_to_f16_kernel<<<blks(4096 * 128, 256), 256, 0, stream>>>(nn2_w, nn2w_h, 4096 * 128);

    // 3) edge MLP layer2 via WMMA (all-f16 inputs, 4x N blocking) -> We [E,4096] f16
    {
        int waves = (E / 16) * (4096 / 64);
        wmma_gemm_f16_kernel<128><<<blks(waves, 4), 128, 0, stream>>>(
            hidden_h, nn2w_h, nn2_b, We, E, 4096);
    }

    // 4) transpose root_w so GEMM sees it as [N,K]
    transpose64_kernel<<<blks(4096, 256), 256, 0, stream>>>(root_w, rootT);

    // 5) degree -> deginv
    fill_f32_kernel<<<blks(N, 256), 256, 0, stream>>>(degbuf, 0.0f, N);
    deg_count_kernel<<<blks(E, 256), 256, 0, stream>>>(ei, degbuf);
    deginv_kernel<<<blks(N, 256), 256, 0, stream>>>(degbuf, deginv);

    // 6) three NNConv + GRU steps
    for (int step = 0; step < 3; ++step) {
        fill_f32_kernel<<<blks(N * 64, 256), 256, 0, stream>>>(agg, 0.0f, N * 64);
        edge_msg_kernel<<<blks(E, 8), 256, 0, stream>>>(We, h, ei, agg);

        // m = relu(out @ root_w + agg/deg + conv_b)
        {
            int waves = (N / 16) * (64 / 16);
            wmma_gemm_f32_kernel<64><<<blks(waves, 4), 128, 0, stream>>>(
                h, rootT, conv_b, agg, deginv, mbuf, N, 64, 1);
        }
        // gx = m @ gru_w_ih^T + b_ih ; gh = h @ gru_w_hh^T + b_hh
        {
            int waves = (N / 16) * (192 / 16);
            wmma_gemm_f32_kernel<64><<<blks(waves, 4), 128, 0, stream>>>(
                mbuf, gru_w_ih, gru_b_ih, nullptr, nullptr, gx, N, 192, 0);
            wmma_gemm_f32_kernel<64><<<blks(waves, 4), 128, 0, stream>>>(
                h, gru_w_hh, gru_b_hh, nullptr, nullptr, gh, N, 192, 0);
        }
        gru_kernel<<<blks(N * 64, 256), 256, 0, stream>>>(gx, gh, h);
    }

    // 7) Set2Set (3 steps)
    fill_f32_kernel<<<blks(B * 128, 256), 256, 0, stream>>>(qstar, 0.0f, B * 128);
    fill_f32_kernel<<<blks(B * 64, 256), 256, 0, stream>>>(hs, 0.0f, B * 64);
    fill_f32_kernel<<<blks(B * 64, 256), 256, 0, stream>>>(cs, 0.0f, B * 64);
    for (int it = 0; it < 3; ++it) {
        lstm_cell_kernel<<<B, 64, 0, stream>>>(qstar, hs, cs,
                                               lstm_w_ih, lstm_w_hh, lstm_b_ih, lstm_b_hh);
        fill_u32_kernel<<<blks(B, 256), 256, 0, stream>>>(emaxu, 0u, B);
        fill_f32_kernel<<<blks(B, 256), 256, 0, stream>>>(asum, 0.0f, B);
        fill_f32_kernel<<<blks(B * 64, 256), 256, 0, stream>>>(rread, 0.0f, B * 64);
        attn_dot_kernel<<<blks(N, 256), 256, 0, stream>>>(h, hs, batch, ebuf, emaxu);
        attn_exp_kernel<<<blks(N, 256), 256, 0, stream>>>(ebuf, emaxu, batch, asum);
        attn_scatter_kernel<<<blks(N * 64, 256), 256, 0, stream>>>(ebuf, asum, h, batch, rread);
        qstar_update_kernel<<<blks(B * 64, 256), 256, 0, stream>>>(hs, rread, qstar);
    }

    // 8) readout -> y at d_out[0..B)
    float* yout = (float*)d_out;
    readout_kernel<<<B, 64, 0, stream>>>(qstar, lin1_w, lin1_b, lin2_w, lin2_b, yout);

    // 9) per_atom_out -> d_out[B ..)
    hipMemcpyAsync(yout + B, h, (size_t)N * 64 * sizeof(float),
                   hipMemcpyDeviceToDevice, stream);
}
